// OptimizedGlobalPointer_63513976373352
// MI455X (gfx1250) — compile-verified, matrix operand order checked by reading, not run on hardware
//
#include <hip/hip_runtime.h>
#include <hip/hip_bf16.h>
#include <math.h>

// Problem constants (from reference)
#define HIDDEN    1024
#define HEADS     12
#define HEAD_DIM  64
#define BATCH     16
#define SEQ       512
#define PROJ_COLS (HEADS * HEAD_DIM * 2)   // 1536
#define ROWS      (BATCH * SEQ)            // 8192
#define NEGV      (-1000000000000.0f)

typedef __attribute__((ext_vector_type(16))) _Float16 v16h;
typedef __attribute__((ext_vector_type(8)))  _Float16 v8h;
typedef __attribute__((ext_vector_type(4)))  _Float16 v4h;
typedef __attribute__((ext_vector_type(8)))  float    v8f;

static __device__ __forceinline__ v16h cat16(v8h lo, v8h hi) {
    return __builtin_shufflevector(lo, hi, 0,1,2,3,4,5,6,7,8,9,10,11,12,13,14,15);
}

// ---------------------------------------------------------------------------
// 1) hidden_states f32 -> f16 (4 elems / thread, float4 loads)
// ---------------------------------------------------------------------------
__global__ void k_cvt_hs(const float* __restrict__ hs, _Float16* __restrict__ hsH) {
    int i = (blockIdx.x * blockDim.x + threadIdx.x) * 4;  // exact: 8192*256*4 = ROWS*HIDDEN
    float4 v = *(const float4*)(hs + i);
    v4h o;
    o.x = (_Float16)v.x; o.y = (_Float16)v.y; o.z = (_Float16)v.z; o.w = (_Float16)v.w;
    *(v4h*)(hsH + i) = o;
}

// ---------------------------------------------------------------------------
// 2) W (1024 x 1536, f32, row-major) -> Wt (1536 x 1024, f16) via LDS tile
// ---------------------------------------------------------------------------
__global__ void k_twt(const float* __restrict__ W, _Float16* __restrict__ Wt) {
    __shared__ float tile[32][33];
    const int c0 = blockIdx.x * 32;   // column of W (0..1535)
    const int k0 = blockIdx.y * 32;   // row of W    (0..1023)
    const int tx = threadIdx.x;       // 0..31
    const int ty = threadIdx.y;       // 0..7
    #pragma unroll
    for (int j = 0; j < 32; j += 8)
        tile[ty + j][tx] = W[(size_t)(k0 + ty + j) * PROJ_COLS + (c0 + tx)];
    __syncthreads();
    #pragma unroll
    for (int j = 0; j < 32; j += 8)
        Wt[(size_t)(c0 + ty + j) * HIDDEN + (k0 + tx)] = (_Float16)tile[tx][ty + j];
}

// ---------------------------------------------------------------------------
// 3) proj = hsH @ Wt^T + bias   (8192 x 1536, K = 1024)
//    one 16x16 output tile per wave, v_wmma_f32_16x16x32_f16 x 32
// ---------------------------------------------------------------------------
__global__ void k_gemm1(const _Float16* __restrict__ hsH,
                        const _Float16* __restrict__ Wt,
                        const float*    __restrict__ bias,
                        float*          __restrict__ proj) {
    const int warp  = threadIdx.x >> 5;
    const int lane  = threadIdx.x & 31;
    const int tile  = blockIdx.x * 8 + warp;           // 49152 tiles total
    const int tM    = tile / (PROJ_COLS / 16);         // 0..511
    const int tC    = tile % (PROJ_COLS / 16);         // 0..95
    const int half  = lane >> 4;
    const int l15   = lane & 15;
    const int kbase = half * 8;                        // ISA 16-bit A/B lane K-chunking

    const _Float16* arow = hsH + (size_t)(tM * 16 + l15) * HIDDEN;
    const _Float16* brow = Wt  + (size_t)(tC * 16 + l15) * HIDDEN;

    v8f c = {};
    #pragma unroll 4
    for (int kk = 0; kk < HIDDEN; kk += 32) {
        v16h a = cat16(*(const v8h*)(arow + kk + kbase),
                       *(const v8h*)(arow + kk + kbase + 16));
        v16h b = cat16(*(const v8h*)(brow + kk + kbase),
                       *(const v8h*)(brow + kk + kbase + 16));
        c = __builtin_amdgcn_wmma_f32_16x16x32_f16(false, a, false, b,
                                                   (short)0, c, false, false);
    }

    const int   col = tC * 16 + l15;                   // C/D: col = lane%16
    const float bv  = bias[col];
    const int   rowbase = tM * 16 + half * 8;          // C/D: row = vgpr + 8*(lane/16)
    #pragma unroll
    for (int r = 0; r < 8; ++r)
        proj[(size_t)(rowbase + r) * PROJ_COLS + col] = c[r] + bv;
}

// ---------------------------------------------------------------------------
// 4) RoPE + q-scaling; write q,k as f16 in [b, h, n, d] (d contiguous)
//    one thread per (b, n, h, pair i) handles (2i, 2i+1) of both q and k
// ---------------------------------------------------------------------------
__global__ void k_rope(const float* __restrict__ proj,
                       _Float16* __restrict__ qH,
                       _Float16* __restrict__ kH) {
    const int idx = blockIdx.x * blockDim.x + threadIdx.x;  // 16*512*12*32 = 3145728
    const int i   = idx & 31;
    int t = idx >> 5;
    const int h = t % HEADS;  t /= HEADS;
    const int n = t % SEQ;
    const int b = t / SEQ;

    const float inv_freq = __powf(10000.0f, -(float)(2 * i) / (float)HEAD_DIM);
    const float ang = (float)n * inv_freq;
    float s, co;
    __sincosf(ang, &s, &co);

    const float* p = proj + (size_t)(b * SEQ + n) * PROJ_COLS + h * (2 * HEAD_DIM);
    const float qe = p[2 * i],            qo = p[2 * i + 1];
    const float ke = p[HEAD_DIM + 2 * i], ko = p[HEAD_DIM + 2 * i + 1];
    const float scale = 0.125f;  // 1/sqrt(64)

    const size_t o = (size_t)((b * HEADS + h) * SEQ + n) * HEAD_DIM + 2 * i;
    qH[o]     = (_Float16)((qe * co - qo * s) * scale);
    qH[o + 1] = (_Float16)((qo * co + qe * s) * scale);
    kH[o]     = (_Float16)(ke * co - ko * s);
    kH[o + 1] = (_Float16)(ko * co + ke * s);
}

// ---------------------------------------------------------------------------
// 5) scores[b,h] = q @ k^T (512x512, K=64) + key/diag mask + sigmoid
//    one 16x16 tile per wave, 2 WMMAs
// ---------------------------------------------------------------------------
__global__ void k_scores(const _Float16* __restrict__ qH,
                         const _Float16* __restrict__ kH,
                         const int*      __restrict__ aw,
                         float*          __restrict__ out) {
    const int warp = threadIdx.x >> 5;
    const int lane = threadIdx.x & 31;
    const int tile = blockIdx.x * 8 + warp;     // 196608 tiles total
    const int bh   = tile >> 10;                // (SEQ/16)^2 = 1024 tiles per (b,h)
    const int rem  = tile & 1023;
    const int tn   = rem >> 5;
    const int tm   = rem & 31;
    const int b    = bh / HEADS;
    const int half = lane >> 4;
    const int l15  = lane & 15;
    const int kbase = half * 8;

    const _Float16* qrow = qH + ((size_t)bh * SEQ + tn * 16 + l15) * HEAD_DIM;
    const _Float16* krow = kH + ((size_t)bh * SEQ + tm * 16 + l15) * HEAD_DIM;

    v8f c = {};
    #pragma unroll
    for (int kk = 0; kk < HEAD_DIM; kk += 32) {
        v16h a  = cat16(*(const v8h*)(qrow + kk + kbase),
                        *(const v8h*)(qrow + kk + kbase + 16));
        v16h bm = cat16(*(const v8h*)(krow + kk + kbase),
                        *(const v8h*)(krow + kk + kbase + 16));
        c = __builtin_amdgcn_wmma_f32_16x16x32_f16(false, a, false, bm,
                                                   (short)0, c, false, false);
    }

    const int  m     = tm * 16 + l15;
    const bool kmask = (aw[b * SEQ + m] == 0);
    const int  nbase = tn * 16 + half * 8;
    #pragma unroll
    for (int r = 0; r < 8; ++r) {
        const int n = nbase + r;
        // sigmoid(NEG) == 0 exactly in f32
        const float v = (kmask || (n == m)) ? 0.0f
                                            : 1.0f / (1.0f + __expf(-c[r]));
        out[((size_t)bh * SEQ + n) * SEQ + m] = v;
    }
}

// ---------------------------------------------------------------------------
// Launch
// ---------------------------------------------------------------------------
extern "C" void kernel_launch(void* const* d_in, const int* in_sizes, int n_in,
                              void* d_out, int out_size, void* d_ws, size_t ws_size,
                              hipStream_t stream) {
    const float* hs   = (const float*)d_in[0];  // (16, 512, 1024) f32
    const int*   aw   = (const int*)  d_in[1];  // (16, 512) i32
    const float* W    = (const float*)d_in[2];  // (1024, 1536) f32
    const float* bias = (const float*)d_in[3];  // (1536,) f32
    float*       out  = (float*)d_out;          // (16, 12, 512, 512) f32

    char* ws = (char*)d_ws;
    const size_t sz_hsH  = (size_t)ROWS * HIDDEN * sizeof(_Float16);      // 16.8 MB
    const size_t sz_WtH  = (size_t)PROJ_COLS * HIDDEN * sizeof(_Float16); //  3.1 MB
    const size_t sz_proj = (size_t)ROWS * PROJ_COLS * sizeof(float);      // 50.3 MB
    const size_t sz_qk   = (size_t)BATCH * HEADS * SEQ * HEAD_DIM * sizeof(_Float16); // 12.6 MB

    _Float16* hsH  = (_Float16*)(ws);
    _Float16* WtH  = (_Float16*)(ws + sz_hsH);
    float*    proj = (float*)   (ws + sz_hsH + sz_WtH);
    _Float16* qH   = (_Float16*)(ws + sz_hsH + sz_WtH + sz_proj);
    _Float16* kH   = (_Float16*)(ws + sz_hsH + sz_WtH + sz_proj + sz_qk);

    // 1) convert hidden_states to f16: 8388608 elems / 4 per thread / 256
    k_cvt_hs<<<8192, 256, 0, stream>>>(hs, hsH);

    // 2) transpose+convert W: tiles of 32x32
    k_twt<<<dim3(PROJ_COLS / 32, HIDDEN / 32), dim3(32, 8), 0, stream>>>(W, WtH);

    // 3) projection GEMM: 512*96 = 49152 tiles, 8 waves/block
    k_gemm1<<<49152 / 8, 256, 0, stream>>>(hsH, WtH, bias, proj);

    // 4) RoPE: 16*512*12*32 threads
    k_rope<<<(BATCH * SEQ * HEADS * 32) / 256, 256, 0, stream>>>(proj, qH, kH);

    // 5) scores: 16*12*32*32 = 196608 tiles, 8 waves/block
    k_scores<<<196608 / 8, 256, 0, stream>>>(qH, kH, aw, out);
}
